// Attention_66005057405770
// MI455X (gfx1250) — matrix operand order repaired and checked by reading, not compiled, not run on hardware
//
#include <hip/hip_runtime.h>
#include <hip/hip_bf16.h>

// ---------------------------------------------------------------------------
// Types / helpers for CDNA5 WMMA (wave32)
// ---------------------------------------------------------------------------
typedef __attribute__((ext_vector_type(16))) __bf16 v16bf;
typedef __attribute__((ext_vector_type(2)))  __bf16 v2bf;
typedef __attribute__((ext_vector_type(8)))  float  v8f;
typedef __attribute__((ext_vector_type(2)))  float  v2f;

union BF16x16 {
  v16bf v;
  unsigned short s[16];
  unsigned u[8];
  uint4 q[2];
};

// packed f32x2 -> bf16x2 (native v_cvt_pk_bf16_f32 on gfx1250)
__device__ __forceinline__ unsigned pk2bf(float a, float b) {
  v2f f; f.x = a; f.y = b;
  v2bf h = __builtin_convertvector(f, v2bf);
  union { v2bf v; unsigned u; } x; x.v = h;
  return x.u;
}
__device__ __forceinline__ unsigned short f2bf(float a) {
  union { __bf16 h; unsigned short s; } x; x.h = (__bf16)a;
  return x.s;
}

__device__ __forceinline__ v8f wmma_bf16(v16bf a, v16bf b, v8f c) {
  // D = A(16x32 bf16) * B(32x16 bf16) + C(16x16 f32)
  return __builtin_amdgcn_wmma_f32_16x16x32_bf16(
      false, a, false, b, (short)0, c, false, false);
}

#define BATCH 8
#define SEQ   1024
#define EMB   768
#define HEADS 12
#define DHEAD 64
#define E3    2304

// Stage a 32(K) x 128(N) f32 tile of W into LDS transposed+converted:
// Bs[n][k] = bf16(W[k0+k][n0+n]).  256 threads, 16 floats each.
__device__ __forceinline__ void stage_b(unsigned short (*__restrict__ Bs)[40],
                                        const float* __restrict__ W, int ldw,
                                        int n0, int k0, int tid) {
  const int k  = tid >> 3;          // 0..31
  const int ns = (tid & 7) * 16;    // 0,16,..,112
  const float* wp = W + (size_t)(k0 + k) * ldw + n0 + ns;
  float4 f0 = *(const float4*)(wp + 0);
  float4 f1 = *(const float4*)(wp + 4);
  float4 f2 = *(const float4*)(wp + 8);
  float4 f3 = *(const float4*)(wp + 12);
  Bs[ns +  0][k] = f2bf(f0.x); Bs[ns +  1][k] = f2bf(f0.y);
  Bs[ns +  2][k] = f2bf(f0.z); Bs[ns +  3][k] = f2bf(f0.w);
  Bs[ns +  4][k] = f2bf(f1.x); Bs[ns +  5][k] = f2bf(f1.y);
  Bs[ns +  6][k] = f2bf(f1.z); Bs[ns +  7][k] = f2bf(f1.w);
  Bs[ns +  8][k] = f2bf(f2.x); Bs[ns +  9][k] = f2bf(f2.y);
  Bs[ns + 10][k] = f2bf(f2.z); Bs[ns + 11][k] = f2bf(f2.w);
  Bs[ns + 12][k] = f2bf(f3.x); Bs[ns + 13][k] = f2bf(f3.y);
  Bs[ns + 14][k] = f2bf(f3.z); Bs[ns + 15][k] = f2bf(f3.w);
}

// ---------------------------------------------------------------------------
// Kernel 1: QKV = X @ W1 + b1 -> Q[B,H,T,64], K[B,H,T,64], Vt[B,H,64,T] bf16.
// Block tile 64(M) x 128(N), 8 waves (4M x 2N), 4 WMMA tiles per wave,
// double-buffered LDS (one barrier per k-step).
// ---------------------------------------------------------------------------
__global__ __launch_bounds__(256) void qkv_gemm(
    const float* __restrict__ X, const float* __restrict__ W1,
    const float* __restrict__ b1,
    unsigned short* __restrict__ Qb, unsigned short* __restrict__ Kb,
    unsigned short* __restrict__ Vt)
{
  __shared__ __align__(16) unsigned short Bs[2][128][40];

  const int tid  = threadIdx.x;
  const int lane = tid & 31;
  const int w    = tid >> 5;
  const int half = lane >> 4;
  const int l15  = lane & 15;
  const int m0   = blockIdx.y * 64;
  const int n0   = blockIdx.x * 128;
  const int wm   = (w & 3) * 16;
  const int wn   = (w >> 2) * 64;

  v8f acc[4] = {};
  const int row = m0 + wm + l15;

  stage_b(Bs[0], W1, E3, n0, 0, tid);
  __syncthreads();

  for (int k0 = 0; k0 < EMB; k0 += 32) {
    const int cur = (k0 >> 5) & 1;
    if (k0 + 32 < EMB) stage_b(Bs[cur ^ 1], W1, E3, n0, k0 + 32, tid);

    // ---- A operand direct from global (f32 -> packed bf16), A layout ------
    BF16x16 A;
    {
      const float* ap = X + (size_t)row * EMB + k0 + half * 8;
      float4 a0 = *(const float4*)(ap + 0);
      float4 a1 = *(const float4*)(ap + 4);
      float4 a2 = *(const float4*)(ap + 16);
      float4 a3 = *(const float4*)(ap + 20);
      A.u[0] = pk2bf(a0.x, a0.y); A.u[1] = pk2bf(a0.z, a0.w);
      A.u[2] = pk2bf(a1.x, a1.y); A.u[3] = pk2bf(a1.z, a1.w);
      A.u[4] = pk2bf(a2.x, a2.y); A.u[5] = pk2bf(a2.z, a2.w);
      A.u[6] = pk2bf(a3.x, a3.y); A.u[7] = pk2bf(a3.z, a3.w);
    }

    // ---- preload 4 B operands, then 4 back-to-back WMMAs ------------------
    BF16x16 Bt[4];
#pragma unroll
    for (int nt = 0; nt < 4; ++nt) {
      const uint4* bp = (const uint4*)&Bs[cur][wn + nt * 16 + l15][half * 8];
      Bt[nt].q[0] = bp[0]; Bt[nt].q[1] = bp[2];
    }
#pragma unroll
    for (int nt = 0; nt < 4; ++nt)
      acc[nt] = wmma_bf16(A.v, Bt[nt].v, acc[nt]);

    __syncthreads();
  }

  // ---- epilogue: bias + scatter; sel/head are uniform per 16-wide tile ----
#pragma unroll
  for (int nt = 0; nt < 4; ++nt) {
    const int nbase = n0 + wn + nt * 16;     // scalar
    const int sel   = nbase / EMB;           // 0=Q 1=K 2=V
    const int eb    = nbase % EMB;
    const int hh    = eb >> 6;               // head
    const int d     = (eb & 63) + l15;
    const float bias = b1[nbase + l15];
    if (sel == 0) {
#pragma unroll
      for (int r = 0; r < 8; ++r) {
        const int m = m0 + wm + half * 8 + r;
        const size_t bh = (size_t)((m >> 10) * HEADS + hh);
        Qb[(bh * SEQ + (m & 1023)) * DHEAD + d] = f2bf(acc[nt][r] + bias);
      }
    } else if (sel == 1) {
#pragma unroll
      for (int r = 0; r < 8; ++r) {
        const int m = m0 + wm + half * 8 + r;
        const size_t bh = (size_t)((m >> 10) * HEADS + hh);
        Kb[(bh * SEQ + (m & 1023)) * DHEAD + d] = f2bf(acc[nt][r] + bias);
      }
    } else {
#pragma unroll
      for (int r = 0; r < 8; ++r) {
        const int m = m0 + wm + half * 8 + r;
        const size_t bh = (size_t)((m >> 10) * HEADS + hh);
        Vt[(bh * DHEAD + d) * SEQ + (m & 1023)] = f2bf(acc[nt][r] + bias);
      }
    }
  }
}

// ---------------------------------------------------------------------------
// Kernel 2: causal flash attention.  grid = (T/64, B*H), 4 waves per block,
// each wave owns 16 queries; 32-key tiles; online softmax; WMMA both matmuls.
// ---------------------------------------------------------------------------
__global__ __launch_bounds__(128) void attn_kernel(
    const unsigned short* __restrict__ Qb, const unsigned short* __restrict__ Kb,
    const unsigned short* __restrict__ Vt, unsigned short* __restrict__ Zb)
{
  __shared__ __align__(16) unsigned short Pl[4][32][16];  // [wave][key][q]

  const int tid  = threadIdx.x;
  const int lane = tid & 31;
  const int w    = tid >> 5;
  const int half = lane >> 4;
  const int l15  = lane & 15;
  const int bh   = blockIdx.y;
  const int b    = bh / HEADS, h = bh % HEADS;
  const int qb   = blockIdx.x * 64 + w * 16;

  const unsigned short* Q = Qb + (size_t)bh * SEQ * DHEAD;
  const unsigned short* K = Kb + (size_t)bh * SEQ * DHEAD;
  const unsigned short* V = Vt + (size_t)bh * DHEAD * SEQ;

  // Q A-operands (persistent): rows q = qb + l15, Dh=64 -> two K=32 operands
  BF16x16 A0, A1;
  {
    const uint4* qp = (const uint4*)(Q + (size_t)(qb + l15) * DHEAD);
    A0.q[0] = qp[half];
    A0.q[1] = qp[2 + half];
    A1.q[0] = qp[4 + half];
    A1.q[1] = qp[6 + half];
  }

  v8f zacc[4] = {};
  float mrow[8], lrow[8];
#pragma unroll
  for (int r = 0; r < 8; ++r) { mrow[r] = -3.0e30f; lrow[r] = 0.0f; }

  const float scale = 0.125f;     // 1/sqrt(64)
  const int jend = qb + 16;

  for (int j = 0; j < jend; j += 32) {
    // prefetch next key/value tiles (global_prefetch_b8)
    if (j + 32 < jend) {
      __builtin_prefetch(K + (size_t)(j + 32 + lane) * DHEAD, 0, 0);
      __builtin_prefetch(V + (size_t)lane * SEQ + j + 32, 0, 0);
      __builtin_prefetch(V + (size_t)(lane + 32) * SEQ + j + 32, 0, 0);
    }

    // ---- scores: S[16q x 32k] via 4 WMMAs --------------------------------
    BF16x16 B00, B01, B10, B11;
    {
      const uint4* kp;
      kp = (const uint4*)(K + (size_t)(j + l15) * DHEAD + half * 8);
      B00.q[0] = kp[0]; B00.q[1] = kp[2];
      kp = (const uint4*)(K + (size_t)(j + l15) * DHEAD + 32 + half * 8);
      B01.q[0] = kp[0]; B01.q[1] = kp[2];
      kp = (const uint4*)(K + (size_t)(j + 16 + l15) * DHEAD + half * 8);
      B10.q[0] = kp[0]; B10.q[1] = kp[2];
      kp = (const uint4*)(K + (size_t)(j + 16 + l15) * DHEAD + 32 + half * 8);
      B11.q[0] = kp[0]; B11.q[1] = kp[2];
    }
    v8f s0 = {}; s0 = wmma_bf16(A0.v, B00.v, s0); s0 = wmma_bf16(A1.v, B01.v, s0);
    v8f s1 = {}; s1 = wmma_bf16(A0.v, B10.v, s1); s1 = wmma_bf16(A1.v, B11.v, s1);

    // ---- causal mask + online softmax ------------------------------------
    BF16x16 P0p, P1p;
#pragma unroll
    for (int r = 0; r < 8; ++r) {
      const int q  = qb + half * 8 + r;
      const int k0 = j + l15, k1 = k0 + 16;
      float v0 = (k0 <= q) ? s0[r] * scale : -3.0e30f;
      float v1 = (k1 <= q) ? s1[r] * scale : -3.0e30f;
      float mx = fmaxf(v0, v1);
#pragma unroll
      for (int off = 1; off < 16; off <<= 1)
        mx = fmaxf(mx, __shfl_xor(mx, off, 32));
      const float nm    = fmaxf(mrow[r], mx);
      const float alpha = __expf(mrow[r] - nm);
      mrow[r] = nm;
      const float p0 = __expf(v0 - nm);
      const float p1 = __expf(v1 - nm);
      float rs = p0 + p1;
#pragma unroll
      for (int off = 1; off < 16; off <<= 1)
        rs += __shfl_xor(rs, off, 32);
      lrow[r] = lrow[r] * alpha + rs;
#pragma unroll
      for (int c = 0; c < 4; ++c) zacc[c][r] *= alpha;
      P0p.s[r] = f2bf(p0);
      P1p.s[r] = f2bf(p1);
    }

    // ---- C-layout -> A-layout for P via per-wave LDS staging -------------
    *(uint4*)&Pl[w][l15][half * 8]      = P0p.q[0];
    *(uint4*)&Pl[w][16 + l15][half * 8] = P1p.q[0];
    BF16x16 PA;
#pragma unroll
    for (int e = 0; e < 8; ++e) PA.s[e]     = Pl[w][half * 8 + e][l15];
#pragma unroll
    for (int e = 0; e < 8; ++e) PA.s[8 + e] = Pl[w][16 + half * 8 + e][l15];

    // ---- accumulate P @ V ------------------------------------------------
#pragma unroll
    for (int c = 0; c < 4; ++c) {
      BF16x16 Bv;
      const uint4* vp =
          (const uint4*)(V + (size_t)(c * 16 + l15) * SEQ + j + half * 8);
      Bv.q[0] = vp[0]; Bv.q[1] = vp[2];
      zacc[c] = wmma_bf16(PA.v, Bv.v, zacc[c]);
    }
  }

  // ---- normalize and store Z[B,T,E] bf16 ---------------------------------
#pragma unroll
  for (int r = 0; r < 8; ++r) {
    const int q = qb + half * 8 + r;
    const float inv = 1.0f / lrow[r];
    const size_t base = ((size_t)b * SEQ + q) * EMB + h * DHEAD;
#pragma unroll
    for (int c = 0; c < 4; ++c)
      Zb[base + c * 16 + l15] = f2bf(zacc[c][r] * inv);
  }
}

// ---------------------------------------------------------------------------
// Kernel 3: Out = Z @ W2 + b2 (Z bf16 in ws, Out f32).
// Block tile 64(M) x 128(N), 4 WMMA tiles per wave, double-buffered LDS.
// ---------------------------------------------------------------------------
__global__ __launch_bounds__(256) void out_gemm(
    const unsigned short* __restrict__ Zb, const float* __restrict__ W2,
    const float* __restrict__ b2, float* __restrict__ Out)
{
  __shared__ __align__(16) unsigned short Bs[2][128][40];

  const int tid  = threadIdx.x;
  const int lane = tid & 31;
  const int w    = tid >> 5;
  const int half = lane >> 4;
  const int l15  = lane & 15;
  const int m0   = blockIdx.y * 64;
  const int n0   = blockIdx.x * 128;
  const int wm   = (w & 3) * 16;
  const int wn   = (w >> 2) * 64;

  v8f acc[4] = {};
  const int row = m0 + wm + l15;

  stage_b(Bs[0], W2, EMB, n0, 0, tid);
  __syncthreads();

  for (int k0 = 0; k0 < EMB; k0 += 32) {
    const int cur = (k0 >> 5) & 1;
    if (k0 + 32 < EMB) stage_b(Bs[cur ^ 1], W2, EMB, n0, k0 + 32, tid);

    BF16x16 A;   // Z already bf16: direct 16B loads in A layout
    const uint4* ap = (const uint4*)(Zb + (size_t)row * EMB + k0);
    A.q[0] = ap[half];
    A.q[1] = ap[2 + half];

    BF16x16 Bt[4];
#pragma unroll
    for (int nt = 0; nt < 4; ++nt) {
      const uint4* bp = (const uint4*)&Bs[cur][wn + nt * 16 + l15][half * 8];
      Bt[nt].q[0] = bp[0]; Bt[nt].q[1] = bp[2];
    }
#pragma unroll
    for (int nt = 0; nt < 4; ++nt)
      acc[nt] = wmma_bf16(A.v, Bt[nt].v, acc[nt]);

    __syncthreads();
  }

#pragma unroll
  for (int nt = 0; nt < 4; ++nt) {
    const int n = n0 + wn + nt * 16 + l15;
    const float bias = b2[n];
#pragma unroll
    for (int r = 0; r < 8; ++r) {
      const int m = m0 + wm + half * 8 + r;
      Out[(size_t)m * EMB + n] = acc[nt][r] + bias;
    }
  }
}

// ---------------------------------------------------------------------------
extern "C" void kernel_launch(void* const* d_in, const int* in_sizes, int n_in,
                              void* d_out, int out_size, void* d_ws, size_t ws_size,
                              hipStream_t stream) {
  (void)in_sizes; (void)n_in; (void)out_size; (void)ws_size;
  const float* X  = (const float*)d_in[0];
  const float* W1 = (const float*)d_in[1];
  const float* b1 = (const float*)d_in[2];
  const float* W2 = (const float*)d_in[3];
  const float* b2 = (const float*)d_in[4];
  float* Out = (float*)d_out;

  const size_t QKV_ELEMS = (size_t)BATCH * HEADS * SEQ * DHEAD;  // 6,291,456
  unsigned short* Qb = (unsigned short*)d_ws;
  unsigned short* Kb = Qb + QKV_ELEMS;
  unsigned short* Vt = Kb + QKV_ELEMS;
  unsigned short* Zb = Vt + QKV_ELEMS;   // [B,T,E] bf16

  qkv_gemm<<<dim3(E3 / 128, (BATCH * SEQ) / 64), 256, 0, stream>>>(
      X, W1, b1, Qb, Kb, Vt);
  attn_kernel<<<dim3(SEQ / 64, BATCH * HEADS), 128, 0, stream>>>(
      Qb, Kb, Vt, Zb);
  out_gemm<<<dim3(EMB / 128, (BATCH * SEQ) / 64), 256, 0, stream>>>(
      Zb, W2, b2, Out);
}
